// CazzyAporbo_35862976921712
// MI455X (gfx1250) — matrix operand order repaired and checked by reading, not compiled, not run on hardware
//
#include <hip/hip_runtime.h>
#include <hip/hip_bf16.h>

// ---------------------------------------------------------------------------
// MI455X (gfx1250) implementation.
//  - All large GEMMs: v_wmma_f32_16x16x32_bf16, fp32 weights converted to bf16
//    while staging to LDS (model is L2-resident: ~75MB weights << 192MB L2).
//  - LDS tiles are stored pre-swizzled into WMMA register order so each lane
//    loads its whole fragment with 2x ds_load_b128 (no scalar u16 assembly).
//  - Attention fused per (b,h,16-row tile) in one wave32 using WMMA for both
//    QK^T (K=16 padded to 32) and P·V, softmax via intra-half shfl_xor.
// ---------------------------------------------------------------------------

typedef __bf16 bf16_t;
typedef __attribute__((ext_vector_type(16))) __bf16 v16bf;
typedef __attribute__((ext_vector_type(8)))  __bf16 v8bf;
typedef __attribute__((ext_vector_type(8)))  float  v8f;

constexpr int kB = 4, kT = 256, kC = 256, kH = 16, kL = 16;
constexpr int kV = 1400, kFF = 1024, kS = 256, kDH = 16;
constexpr int kNB = 8, kNG = 4, kBioD = 64, kGenD = 128;
constexpr int kKV = kT + kNB + kNG;  // 268 rows in "combined"

// ---- bf16 helpers (round-to-nearest-even, pure bit ops => always compiles) --
__device__ __forceinline__ unsigned short bfbits(float f) {
  unsigned u = __builtin_bit_cast(unsigned, f);
  u += 0x7FFFu + ((u >> 16) & 1u);
  return (unsigned short)(u >> 16);
}
__device__ __forceinline__ bf16_t bfc(float f) {
  unsigned short s = bfbits(f);
  return __builtin_bit_cast(bf16_t, s);
}
__device__ __forceinline__ unsigned bfpack(float lo, float hi) {
  return (unsigned)bfbits(lo) | ((unsigned)bfbits(hi) << 16);
}

__device__ __forceinline__ float actf(float v, int act) {
  switch (act) {
    case 1: return fmaxf(v, 0.f);                                     // relu
    case 2: return 0.5f * v * (1.f + erff(v * 0.70710678118654752f)); // gelu
    case 3: return 1.f / (1.f + __expf(-v));                          // sigmoid
    case 4: return __expf(v);                                         // exp
  }
  return v;
}

// ---------------------------------------------------------------------------
// Tiled WMMA GEMM:  out[M,N] = act(A[M,K] @ W[K,N] + bias) + resid
// 128 threads (4 waves), 64x64 C-tile, K staged 32 at a time.
// LDS layout is fragment-major: [block][lane][element] so the compute side
// reads one contiguous 32B vector per fragment.
//
// ISA fragment mapping (CDNA5 ISA 7.12.2), inverted for the writers:
//   A elem e of lane (half*16+m):  k = (e<8 ? e : e+8) + half*8
//     => for source col c: half=(c>>3)&1, e=(c&7)+((c>=16)?8:0)
//   B elem e of lane (half*16+n):  k = (e<8 ? e : e+8) + half*8
//     => for source row k: half=(k>>3)&1, e=(k&7)+((k>=16)?8:0)
// ---------------------------------------------------------------------------
__global__ __launch_bounds__(128) void gemm_wmma(
    const float* __restrict__ A, int lda,
    const float* __restrict__ W, int ldb,
    const float* __restrict__ bias,
    const float* __restrict__ resid,
    float* __restrict__ out, int ldc,
    int M, int N, int K, int act) {
  __shared__ unsigned short Ash[4][32][16];  // [wave][lane][elem]  4KB
  __shared__ unsigned short Bsh[4][32][16];  // [ntile][lane][elem] 4KB

  const int tid  = threadIdx.x;
  const int lane = tid & 31;
  const int wv   = tid >> 5;  // wave id 0..3 -> 16 output rows each
  const int nIdx = lane & 15;
  const int half = lane >> 4;
  const int m0 = blockIdx.y * 64;
  const int n0 = blockIdx.x * 64;

  v8f acc[4] = {};

  for (int kk = 0; kk < K; kk += 32) {
    // ---- stage A tile (64 rows x 32 k) as packed bf16 pairs ----
    for (int i = tid; i < 64 * 16; i += 128) {
      int r  = i >> 4;
      int cp = (i & 15) << 1;  // even k within tile
      int gr = m0 + r;
      float v0 = 0.f, v1 = 0.f;
      if (gr < M) {
        const float* ap = &A[(size_t)gr * lda + kk + cp];
        v0 = ap[0]; v1 = ap[1];
      }
      int hf = (cp >> 3) & 1;
      int e0 = (cp & 7) + ((cp >> 4) << 3);  // even
      *(unsigned*)&Ash[r >> 4][hf * 16 + (r & 15)][e0] = bfpack(v0, v1);
    }
    // ---- stage B tile (32 k x 64 cols) ----
    for (int i = tid; i < 16 * 64; i += 128) {
      int rp = (i >> 6) << 1;  // even k
      int c  = i & 63;
      int gc = n0 + c;
      float v0 = 0.f, v1 = 0.f;
      if (gc < N) {
        v0 = W[(size_t)(kk + rp) * ldb + gc];
        v1 = W[(size_t)(kk + rp + 1) * ldb + gc];
      }
      *(unsigned*)&Bsh[c >> 4][((rp >> 3) & 1) * 16 + (c & 15)][(rp & 7) + ((rp >> 4) << 3)] = bfpack(v0, v1);
    }
    if (kk + 32 < K) {  // hint next K-slice toward L2 (global_prefetch_b8)
      __builtin_prefetch(&W[(size_t)(kk + 32 + (tid & 31)) * ldb + n0], 0, 1);
    }
    __syncthreads();

    // ---- compute: fragments are contiguous per lane (2x ds_load_b128) ----
    const v16bf afrag = *(const v16bf*)&Ash[wv][lane][0];
#pragma unroll
    for (int nt = 0; nt < 4; ++nt) {
      const v16bf bfrag = *(const v16bf*)&Bsh[nt][lane][0];
      acc[nt] = __builtin_amdgcn_wmma_f32_16x16x32_bf16(
          false, afrag, false, bfrag, (short)0, acc[nt], false, false);
    }
    __syncthreads();
  }

  // Epilogue: C/D layout -> element r of v8f is row (r + 8*half), col nIdx.
#pragma unroll
  for (int nt = 0; nt < 4; ++nt) {
#pragma unroll
    for (int r = 0; r < 8; ++r) {
      int row = m0 + wv * 16 + r + half * 8;
      int col = n0 + nt * 16 + nIdx;
      if (row < M && col < N) {
        float v = acc[nt][r];
        if (bias)  v += bias[col];
        v = actf(v, act);
        if (resid) v += resid[(size_t)row * ldc + col];
        out[(size_t)row * ldc + col] = v;
      }
    }
  }
}

// ---------------------------------------------------------------------------
// Fused attention: one wave32 per (b, h, 16-query-row tile).
// scores = (Q K^T)/4 [+ 0.1*aff gather] -> softmax -> P V, all via WMMA.
// Generic kv length skv (<= 272) for the 268-row cross-attention.
// ---------------------------------------------------------------------------
__global__ __launch_bounds__(32) void attn_wmma(
    const float* __restrict__ qb, int qs,
    const float* __restrict__ kb, int ks,
    const float* __restrict__ vb, int vs,
    int skv, int kvRows,
    float* __restrict__ ob,
    const float* __restrict__ aff, const int* __restrict__ ids,
    float affScale) {
  __shared__ unsigned short Plds[16][288];  // probs bf16, padded for b128 loads

  const int lane = threadIdx.x;
  const int nIdx = lane & 15;
  const int half = lane >> 4;
  const int b = blockIdx.z, h = blockIdx.y;
  const int t0 = blockIdx.x * 16;
  const int ntc = (skv + 15) / 16;    // score tiles  (<= 17)
  const int ntcol = ntc * 16;

  // zero prob-tile tail so P-fragment vector loads never read garbage
  for (int col = ntcol + lane; col < 288; col += 32) {
#pragma unroll
    for (int r = 0; r < 16; ++r) Plds[r][col] = 0;
  }

  // Q A-fragment: lane's dh slice (8 floats, 2x float4); upper half = K pad.
  v16bf qa;
  {
    const float* qrow =
        qb + ((size_t)(b * kT + t0 + nIdx)) * qs + h * kDH + half * 8;
    float4 q0 = *(const float4*)(qrow);
    float4 q1 = *(const float4*)(qrow + 4);
    qa[0] = bfc(q0.x); qa[1] = bfc(q0.y); qa[2] = bfc(q0.z); qa[3] = bfc(q0.w);
    qa[4] = bfc(q1.x); qa[5] = bfc(q1.y); qa[6] = bfc(q1.z); qa[7] = bfc(q1.w);
#pragma unroll
    for (int e = 8; e < 16; ++e) qa[e] = bfc(0.f);
  }

  v8f acc[17];
  for (int nt = 0; nt < ntc; ++nt) {
    // B fragment of K^T: elems 0..7 carry dh rows half*8..half*8+7,
    // elems 8..15 are the zero K-padding rows 16..31.
    v16bf kf;
    int s = nt * 16 + nIdx;
    if (s < skv) {
      const float* krow = kb + ((size_t)(b * kvRows + s)) * ks + h * kDH + half * 8;
      float4 a0 = *(const float4*)(krow);
      float4 a1 = *(const float4*)(krow + 4);
      kf[0]  = bfc(a0.x); kf[1]  = bfc(a0.y); kf[2]  = bfc(a0.z); kf[3]  = bfc(a0.w);
      kf[4]  = bfc(a1.x); kf[5]  = bfc(a1.y); kf[6]  = bfc(a1.z); kf[7]  = bfc(a1.w);
#pragma unroll
      for (int e = 8; e < 16; ++e) kf[e] = bfc(0.f);
    } else {
#pragma unroll
      for (int e = 0; e < 16; ++e) kf[e] = bfc(0.f);
    }
    v8f z = {};
    acc[nt] = __builtin_amdgcn_wmma_f32_16x16x32_bf16(
        false, qa, false, kf, (short)0, z, false, false);
  }

  // scale + affinity gather + kv mask
  for (int nt = 0; nt < ntc; ++nt) {
#pragma unroll
    for (int r = 0; r < 8; ++r) {
      int s = nt * 16 + nIdx;
      float v = acc[nt][r] * 0.25f;  // 1/sqrt(16)
      if (s >= skv) {
        v = -1e30f;
      } else if (aff) {
        int t = t0 + r + half * 8;
        v += affScale * aff[(size_t)ids[b * kT + t] * kV + ids[b * kT + s]];
      }
      acc[nt][r] = v;
    }
  }

  // softmax per row: row = (r, half); cols spread over the 16 lanes of a half
  float sm[8];
#pragma unroll
  for (int r = 0; r < 8; ++r) {
    float m = -1e30f;
    for (int nt = 0; nt < ntc; ++nt) m = fmaxf(m, acc[nt][r]);
    for (int off = 1; off < 16; off <<= 1) m = fmaxf(m, __shfl_xor(m, off, 32));
    float s = 0.f;
    for (int nt = 0; nt < ntc; ++nt) {
      float e2 = __expf(acc[nt][r] - m);
      acc[nt][r] = e2;
      s += e2;
    }
    for (int off = 1; off < 16; off <<= 1) s += __shfl_xor(s, off, 32);
    sm[r] = s;
  }

  // relayout probs C->A via LDS (bf16)
  for (int nt = 0; nt < ntc; ++nt) {
#pragma unroll
    for (int r = 0; r < 8; ++r) {
      Plds[r + half * 8][nt * 16 + nIdx] = bfbits(acc[nt][r] / sm[r]);
    }
  }
  __syncthreads();

  // O = P (16 x skv) @ V (skv x 16)
  v8f oacc = {};
  const int ktc = (skv + 31) / 32;
  for (int kt = 0; kt < ktc; ++kt) {
    // P A-fragment: two contiguous 16B chunks per lane (ds_load_b128 x2)
    const unsigned short* prow = &Plds[nIdx][kt * 32 + half * 8];
    v8bf lo = *(const v8bf*)(prow);
    v8bf hi = *(const v8bf*)(prow + 16);
    v16bf pa;
#pragma unroll
    for (int e = 0; e < 8; ++e) { pa[e] = lo[e]; pa[e + 8] = hi[e]; }
    // V B-fragment: inherently transposed access, scalar gather
    v16bf vf;
#pragma unroll
    for (int e = 0; e < 16; ++e) {
      int s2 = kt * 32 + (e & 7) + ((e >> 3) << 4) + half * 8;
      float v = (s2 < skv)
                    ? vb[((size_t)(b * kvRows + s2)) * vs + h * kDH + nIdx] : 0.f;
      vf[e] = bfc(v);
    }
    oacc = __builtin_amdgcn_wmma_f32_16x16x32_bf16(
        false, pa, false, vf, (short)0, oacc, false, false);
  }
#pragma unroll
  for (int r = 0; r < 8; ++r) {
    int t = t0 + r + half * 8;
    ob[((size_t)(b * kT + t)) * kC + h * kDH + nIdx] = oacc[r];
  }
}

// ---------------------------------------------------------------------------
// Elementwise / small kernels
// ---------------------------------------------------------------------------
__global__ __launch_bounds__(256) void embed_temporal(
    const int* __restrict__ ids, const float* __restrict__ ages,
    const float* __restrict__ emb, const float* __restrict__ freq_w,
    const float* __restrict__ phase, const float* __restrict__ amp,
    const float* __restrict__ rw1, const float* __restrict__ rb1,
    const float* __restrict__ rw2, const float* __restrict__ rb2,
    float* __restrict__ x) {
  int row = blockIdx.x, c = threadIdx.x;
  float an = ages[row] * (1.f / (120.f * 365.25f));
  __shared__ float hid[64];
  if (c < 64) hid[c] = fmaxf(an * rw1[c] + rb1[c], 0.f);
  __syncthreads();
  float a2 = rb2[c];
  for (int j = 0; j < 64; ++j) a2 += hid[j] * rw2[j * kC + c];
  float res = 1.f / (1.f + __expf(-a2));
  int ci = (c < kC / 2) ? c : c - kC / 2;
  float bf2 = __expf(freq_w[ci]) * (1.f / 365.25f);
  float ang = 2.f * 3.14159265358979323846f * an * bf2 + phase[ci];
  float tv = ((c < kC / 2) ? __sinf(ang) : __cosf(ang)) * amp[ci];
  x[(size_t)row * kC + c] = emb[(size_t)ids[row] * kC + c] + tv * res;
}

__global__ __launch_bounds__(256) void ln_kernel(
    const float* __restrict__ xin, float* __restrict__ xout,
    const float* __restrict__ g, const float* __restrict__ be, int do_relu) {
  int row = blockIdx.x, c = threadIdx.x;
  __shared__ float red[256];
  float v = xin[(size_t)row * kC + c];
  red[c] = v; __syncthreads();
  for (int s = 128; s > 0; s >>= 1) { if (c < s) red[c] += red[c + s]; __syncthreads(); }
  float m = red[0] * (1.f / kC); __syncthreads();
  float d = v - m;
  red[c] = d * d; __syncthreads();
  for (int s = 128; s > 0; s >>= 1) { if (c < s) red[c] += red[c + s]; __syncthreads(); }
  float var = red[0] * (1.f / kC);
  float y = d * rsqrtf(var + 1e-5f) * g[c] + be[c];
  if (do_relu) y = fmaxf(y, 0.f);
  xout[(size_t)row * kC + c] = y;
}

__global__ __launch_bounds__(256) void softmax_rows(float* __restrict__ x) {
  int row = blockIdx.x, c = threadIdx.x;
  __shared__ float red[256];
  float v = x[(size_t)row * kS + c];
  red[c] = v; __syncthreads();
  for (int s = 128; s > 0; s >>= 1) { if (c < s) red[c] = fmaxf(red[c], red[c + s]); __syncthreads(); }
  float m = red[0]; __syncthreads();
  float e2 = __expf(v - m);
  red[c] = e2; __syncthreads();
  for (int s = 128; s > 0; s >>= 1) { if (c < s) red[c] += red[c + s]; __syncthreads(); }
  x[(size_t)row * kS + c] = e2 / red[0];
}

__global__ __launch_bounds__(256) void small_gemm(
    const float* __restrict__ A, int lda, const float* __restrict__ W,
    const float* __restrict__ bias, float* __restrict__ out, int ldo,
    int K, int N, int act, int inRowsPerB, int outRowsPerB, int outRowOff) {
  int m = blockIdx.x, n = threadIdx.x;
  if (n >= N) return;
  float s = bias ? bias[n] : 0.f;
  for (int k = 0; k < K; ++k) s += A[(size_t)m * lda + k] * W[(size_t)k * N + n];
  s = actf(s, act);
  int b = m / inRowsPerB, r = m % inRowsPerB;
  out[((size_t)(b * outRowsPerB + outRowOff + r)) * ldo + n] = s;
}

__global__ __launch_bounds__(256) void copy_to_combined(
    const float* __restrict__ x, float* __restrict__ comb) {
  int row = blockIdx.x, c = threadIdx.x;
  int b = row / kT, t = row % kT;
  comb[((size_t)(b * kKV + t)) * kC + c] = x[(size_t)row * kC + c];
}

__global__ __launch_bounds__(256) void mean_combined(
    const float* __restrict__ comb, float* __restrict__ meanc) {
  int b = blockIdx.x, c = threadIdx.x;
  float s = 0.f;
  for (int r = 0; r < kKV; ++r) s += comb[((size_t)(b * kKV + r)) * kC + c];
  meanc[b * kC + c] = s * (1.f / kKV);
}

__global__ __launch_bounds__(256) void concat3(
    const float* __restrict__ x, const float* __restrict__ att,
    const float* __restrict__ meanc, float* __restrict__ out) {
  int row = blockIdx.x, c = threadIdx.x;
  int b = row / kT;
  out[(size_t)row * 3 * kC + c]          = x[(size_t)row * kC + c];
  out[(size_t)row * 3 * kC + kC + c]     = att[(size_t)row * kC + c];
  out[(size_t)row * 3 * kC + 2 * kC + c] = meanc[b * kC + c];
}

__global__ __launch_bounds__(256) void gate_combine(
    float* __restrict__ x, const float* __restrict__ g,
    const float* __restrict__ att) {
  size_t i = (size_t)blockIdx.x * blockDim.x + threadIdx.x;
  float gg = g[i];
  x[i] = gg * x[i] + (1.f - gg) * att[i];
}

__global__ __launch_bounds__(256) void time_head(
    const float* __restrict__ t1, const float* __restrict__ w2,
    const float* __restrict__ b2, float* __restrict__ out, int M) {
  int m = blockIdx.x * blockDim.x + threadIdx.x;
  if (m < M) {
    float s = b2[0];
    for (int j = 0; j < kC / 2; ++j) s += t1[(size_t)m * (kC / 2) + j] * w2[j];
    out[m] = s;
  }
}

// ---------------------------------------------------------------------------
// Host orchestration
// ---------------------------------------------------------------------------
enum {
  IN_IDS = 0, IN_AGES, IN_BIO, IN_GEN,
  P_EMB, P_FREQ, P_PHASE, P_AMP,
  P_RES_W1, P_RES_B1, P_RES_W2, P_RES_B2,
  P_BIO_W1, P_BIO_B1, P_BIO_G, P_BIO_BETA, P_BIO_W2, P_BIO_B2,
  P_GEN_W1, P_GEN_B1, P_GEN_G, P_GEN_BETA, P_GEN_W2, P_GEN_B2,
  P_XIN_W, P_XIN_B, P_XOUT_W, P_XOUT_B, P_GATE_W, P_GATE_B,
  L_AIN_W, L_AIN_B, L_AOUT_W, L_AOUT_B,
  L_GAQ_W, L_GAQ_B, L_GAK_W, L_GAK_B, L_GAV_W, L_GAV_B, L_GAO_W, L_GAO_B,
  L_GA_AFF,
  L_LN1G, L_LN1B, L_LN2G, L_LN2B, L_LN3G, L_LN3B,
  L_FFN_W1, L_FFN_B1, L_FFN_W2, L_FFN_B2,
  L_MEM, L_READ_W, L_READ_B,
  P_UM_W, P_UM_B, P_ULV_W, P_ULV_B, P_DIS_W, P_DIS_B,
  P_TW1, P_TB1, P_TW2, P_TB2
};

static inline void gemm(hipStream_t s, const float* A, int lda,
                        const float* W, int ldb, const float* bias,
                        const float* resid, float* out, int ldc,
                        int M, int N, int K, int act) {
  dim3 g((N + 63) / 64, (M + 63) / 64);
  gemm_wmma<<<g, 128, 0, s>>>(A, lda, W, ldb, bias, resid, out, ldc, M, N, K, act);
}

extern "C" void kernel_launch(void* const* d_in, const int* in_sizes, int n_in,
                              void* d_out, int out_size, void* d_ws, size_t ws_size,
                              hipStream_t stream) {
  (void)in_sizes; (void)n_in; (void)out_size; (void)ws_size;
  auto F = [&](int i) { return (const float*)d_in[i]; };
  const int* ids = (const int*)d_in[IN_IDS];
  const int MT = kB * kT;  // 1024 token rows

  float* ws = (float*)d_ws;
  auto alloc = [&](size_t n) { float* p = ws; ws += n; return p; };

  float* x     = alloc((size_t)MT * kC);
  float* h     = alloc((size_t)MT * kC);
  float* att   = alloc((size_t)MT * kC);
  float* att2  = alloc((size_t)MT * kC);
  float* comb  = alloc((size_t)kB * kKV * kC);
  float* qx    = alloc((size_t)MT * kC);
  float* kvc   = alloc((size_t)kB * kKV * 2 * kC);
  float* meanc = alloc((size_t)kB * kC);
  float* g3    = alloc((size_t)MT * 3 * kC);
  float* gateg = alloc((size_t)MT * kC);
  float* qkv   = alloc((size_t)MT * 3 * kC);
  float* rw    = alloc((size_t)MT * kS);
  float* mid   = alloc((size_t)MT * kFF);
  float* meanb = alloc((size_t)MT * kC);
  float* t1    = alloc((size_t)MT * (kC / 2));
  float* bioh  = alloc((size_t)kB * kNB * kC);
  float* bioh2 = alloc((size_t)kB * kNB * kC);
  float* genh  = alloc((size_t)kB * kNG * kC);
  float* genh2 = alloc((size_t)kB * kNG * kC);

  float* out_logits = (float*)d_out;
  float* out_time   = out_logits + (size_t)MT * kV;
  float* out_var    = out_time + MT;

  // 1. Embedding + temporal encoding
  embed_temporal<<<MT, 256, 0, stream>>>(ids, F(IN_AGES), F(P_EMB), F(P_FREQ),
      F(P_PHASE), F(P_AMP), F(P_RES_W1), F(P_RES_B1), F(P_RES_W2), F(P_RES_B2), x);

  // 2. Bio / genetics encoders -> combined rows [256..268)
  small_gemm<<<kB * kNB, 256, 0, stream>>>(F(IN_BIO), kBioD, F(P_BIO_W1),
      F(P_BIO_B1), bioh, kC, kBioD, kC, 0, kB * kNB, kB * kNB, 0);
  ln_kernel<<<kB * kNB, 256, 0, stream>>>(bioh, bioh2, F(P_BIO_G), F(P_BIO_BETA), 1);
  small_gemm<<<kB * kNB, 256, 0, stream>>>(bioh2, kC, F(P_BIO_W2), F(P_BIO_B2),
      comb, kC, kC, kC, 0, kNB, kKV, kT);
  small_gemm<<<kB * kNG, 256, 0, stream>>>(F(IN_GEN), kGenD, F(P_GEN_W1),
      F(P_GEN_B1), genh, kC, kGenD, kC, 0, kB * kNG, kB * kNG, 0);
  ln_kernel<<<kB * kNG, 256, 0, stream>>>(genh, genh2, F(P_GEN_G), F(P_GEN_BETA), 1);
  small_gemm<<<kB * kNG, 256, 0, stream>>>(genh2, kC, F(P_GEN_W2), F(P_GEN_B2),
      comb, kC, kC, kC, 0, kNG, kKV, kT + kNB);
  copy_to_combined<<<MT, 256, 0, stream>>>(x, comb);

  // 3. Cross-attention (q from x, kv from combined[268])
  gemm(stream, x, kC, F(P_XIN_W), 3 * kC, F(P_XIN_B), nullptr,
       qx, kC, MT, kC, kC, 0);
  gemm(stream, comb, kC, F(P_XIN_W) + kC, 3 * kC, F(P_XIN_B) + kC, nullptr,
       kvc, 2 * kC, kB * kKV, 2 * kC, kC, 0);
  attn_wmma<<<dim3(kT / 16, kH, kB), 32, 0, stream>>>(
      qx, kC, kvc, 2 * kC, kvc + kC, 2 * kC, kKV, kKV, att,
      nullptr, nullptr, 0.f);
  gemm(stream, att, kC, F(P_XOUT_W), kC, F(P_XOUT_B), nullptr,
       att2, kC, MT, kC, kC, 0);

  // 4. Gating
  mean_combined<<<kB, 256, 0, stream>>>(comb, meanc);
  concat3<<<MT, 256, 0, stream>>>(x, att2, meanc, g3);
  gemm(stream, g3, 3 * kC, F(P_GATE_W), kC, F(P_GATE_B), nullptr,
       gateg, kC, MT, kC, 3 * kC, 3);
  gate_combine<<<MT, 256, 0, stream>>>(x, gateg, att2);

  // 5. Transformer layers
  for (int l = 0; l < kL; ++l) {
    const float* ain_w  = F(L_AIN_W)  + (size_t)l * kC * 3 * kC;
    const float* ain_b  = F(L_AIN_B)  + (size_t)l * 3 * kC;
    const float* aout_w = F(L_AOUT_W) + (size_t)l * kC * kC;
    const float* aout_b = F(L_AOUT_B) + (size_t)l * kC;

    // self-attention
    ln_kernel<<<MT, 256, 0, stream>>>(x, h, F(L_LN1G) + l * kC, F(L_LN1B) + l * kC, 0);
    gemm(stream, h, kC, ain_w, 3 * kC, ain_b, nullptr, qkv, 3 * kC, MT, 3 * kC, kC, 0);
    attn_wmma<<<dim3(kT / 16, kH, kB), 32, 0, stream>>>(
        qkv, 3 * kC, qkv + kC, 3 * kC, qkv + 2 * kC, 3 * kC, kT, kT, att,
        nullptr, nullptr, 0.f);
    gemm(stream, att, kC, aout_w, kC, aout_b, x, x, kC, MT, kC, kC, 0);

    // graph attention (with affinity-table bias)
    ln_kernel<<<MT, 256, 0, stream>>>(x, h, F(L_LN2G) + l * kC, F(L_LN2B) + l * kC, 0);
    gemm(stream, h, kC, F(L_GAQ_W) + (size_t)l * kC * kC, kC,
         F(L_GAQ_B) + l * kC, nullptr, qkv, 3 * kC, MT, kC, kC, 0);
    gemm(stream, h, kC, F(L_GAK_W) + (size_t)l * kC * kC, kC,
         F(L_GAK_B) + l * kC, nullptr, qkv + kC, 3 * kC, MT, kC, kC, 0);
    gemm(stream, h, kC, F(L_GAV_W) + (size_t)l * kC * kC, kC,
         F(L_GAV_B) + l * kC, nullptr, qkv + 2 * kC, 3 * kC, MT, kC, kC, 0);
    attn_wmma<<<dim3(kT / 16, kH, kB), 32, 0, stream>>>(
        qkv, 3 * kC, qkv + kC, 3 * kC, qkv + 2 * kC, 3 * kC, kT, kT, att,
        F(L_GA_AFF) + (size_t)l * kV * kV, ids, 0.1f);
    gemm(stream, att, kC, F(L_GAO_W) + (size_t)l * kC * kC, kC,
         F(L_GAO_B) + l * kC, x, x, kC, MT, kC, kC, 0);

    // memory read
    gemm(stream, x, kC, F(L_READ_W) + (size_t)l * kC * kS, kS,
         F(L_READ_B) + l * kS, nullptr, rw, kS, MT, kS, kC, 0);
    softmax_rows<<<MT, 256, 0, stream>>>(rw);
    gemm(stream, rw, kS, F(L_MEM) + (size_t)l * kS * kC, kC,
         nullptr, x, x, kC, MT, kC, kS, 0);

    // FFN (gelu)
    ln_kernel<<<MT, 256, 0, stream>>>(x, h, F(L_LN3G) + l * kC, F(L_LN3B) + l * kC, 0);
    gemm(stream, h, kC, F(L_FFN_W1) + (size_t)l * kC * kFF, kFF,
         F(L_FFN_B1) + l * kFF, nullptr, mid, kFF, MT, kFF, kC, 2);
    gemm(stream, mid, kFF, F(L_FFN_W2) + (size_t)l * kFF * kC, kC,
         F(L_FFN_B2) + l * kC, x, x, kC, MT, kC, kFF, 0);
  }

  // 6. Heads
  gemm(stream, x, kC, F(P_UM_W), kC, F(P_UM_B), nullptr, meanb, kC, MT, kC, kC, 0);
  gemm(stream, x, kC, F(P_ULV_W), kC, F(P_ULV_B), nullptr, out_var, kC, MT, kC, kC, 4);
  gemm(stream, meanb, kC, F(P_DIS_W), kV, F(P_DIS_B), nullptr,
       out_logits, kV, MT, kV, kC, 0);
  gemm(stream, meanb, kC, F(P_TW1), kC / 2, F(P_TB1), nullptr,
       t1, kC / 2, MT, kC / 2, kC, 1);
  time_head<<<(MT + 255) / 256, 256, 0, stream>>>(t1, F(P_TW2), F(P_TB2),
                                                  out_time, MT);
}
